// ScaledDotProductAttention_20280835571925
// MI455X (gfx1250) — matrix operand order repaired and checked by reading, not compiled, not run on hardware
//
#include <hip/hip_runtime.h>
#include <hip/hip_bf16.h>
#include <stdint.h>

// Problem dims
#define BB 4
#define LL 2048
#define DD 1024
#define HH 16
#define DH 64
#define NROW (BB * LL)          // 8192
#define MASKV (-1000000.0f)
#define SCALE 0.125f            // 1/sqrt(64)

typedef _Float16 half8  __attribute__((ext_vector_type(8)));
typedef _Float16 half16 __attribute__((ext_vector_type(16)));
typedef float    f32x8  __attribute__((ext_vector_type(8)));

__device__ __forceinline__ half16 ld2(const _Float16* p0, const _Float16* p1) {
    union { half16 v; half8 h[2]; } u;
    u.h[0] = *reinterpret_cast<const half8*>(p0);
    u.h[1] = *reinterpret_cast<const half8*>(p1);
    return u.v;
}

__device__ __forceinline__ f32x8 wmma_f16(half16 a, half16 b, f32x8 c) {
    // D = A(16x32 f16) * B(32x16 f16) + C(16x16 f32)
    return __builtin_amdgcn_wmma_f32_16x16x32_f16(false, a, false, b, (short)0, c,
                                                  false, false);
}

// ---------------------------------------------------------------------------
// f32 -> f16 elementwise convert
__global__ __launch_bounds__(256) void cvt_f16_kernel(const float* __restrict__ in,
                                                      _Float16* __restrict__ out, int n) {
    int i = blockIdx.x * 256 + threadIdx.x;
    if (i < n) out[i] = (_Float16)in[i];
}

// Transpose 1024x1024 f32 weight -> f16 [n][k]
__global__ __launch_bounds__(256) void wtrans_kernel(const float* __restrict__ W,
                                                     _Float16* __restrict__ Wt) {
    int n = blockIdx.x;                 // output column -> row of Wt
    int k0 = threadIdx.x * 4;
#pragma unroll
    for (int j = 0; j < 4; ++j)
        Wt[(size_t)n * DD + k0 + j] = (_Float16)W[(size_t)(k0 + j) * DD + n];
}

// Repack V: vh[b, n, h*64+d] -> vt[((b*16+h)*64+d)*2048 + n]
__global__ __launch_bounds__(256) void repackv_kernel(const _Float16* __restrict__ vh,
                                                      _Float16* __restrict__ vt) {
    size_t i = (size_t)blockIdx.x * 256 + threadIdx.x;   // over B*L*D = 8M
    int d = (int)(i & 63);
    int h = (int)((i >> 6) & 15);
    int n = (int)((i >> 10) & 2047);
    int b = (int)(i >> 21);
    vt[((size_t)(b * 16 + h) * 64 + d) * (size_t)LL + n] = vh[i];
}

// ---------------------------------------------------------------------------
// GEMM: C[8192 x 1024] = A[8192 x 1024] (f16 row-major) @ Bt (f16, stored [n][k])
// One wave computes a 16(M) x 64(N) tile; K=1024 in steps of 32.
__global__ __launch_bounds__(256) void gemm_kernel(const _Float16* __restrict__ A,
                                                   const _Float16* __restrict__ Bt,
                                                   _Float16* __restrict__ C) {
    int wid   = blockIdx.x * 8 + (threadIdx.x >> 5);   // 8192 waves
    int ntile = wid & 15;                              // 16 tiles of 64 cols
    int mtile = wid >> 4;                              // 512 tiles of 16 rows
    int lane  = threadIdx.x & 31;
    int hi    = lane >> 4;
    int lq    = lane & 15;

    const _Float16* arow = A + ((size_t)mtile * 16 + lq) * DD;
    f32x8 acc[4] = {};

    for (int k0 = 0; k0 < DD; k0 += 32) {
        // A operand: lane half picks 16 contiguous K values {hi*16 .. +15}
        half16 a = ld2(arow + k0 + hi * 16, arow + k0 + hi * 16 + 8);
        __builtin_prefetch(arow + k0 + 64, 0, 1);
#pragma unroll
        for (int sub = 0; sub < 4; ++sub) {
            const _Float16* bcol =
                Bt + ((size_t)ntile * 64 + sub * 16 + lq) * DD + k0 + hi * 16;
            half16 b = ld2(bcol, bcol + 8);   // 16 contiguous K values per lane-half
            acc[sub] = wmma_f16(a, b, acc[sub]);
        }
    }

#pragma unroll
    for (int sub = 0; sub < 4; ++sub)
#pragma unroll
        for (int r = 0; r < 8; ++r)
            C[((size_t)mtile * 16 + r + 8 * hi) * DD + ntile * 64 + sub * 16 + lq] =
                (_Float16)acc[sub][r];
}

// ---------------------------------------------------------------------------
// Flash attention: one wave (one block of 32) per 16-query tile of one (b,h).
// Streams 32 keys per iteration; online softmax in C-layout registers.
__global__ __launch_bounds__(32) void attn_kernel(const _Float16* __restrict__ qh,
                                                  const _Float16* __restrict__ kh,
                                                  const _Float16* __restrict__ vt,
                                                  _Float16* __restrict__ zh) {
    __shared__ __align__(16) _Float16 pbuf[16 * 32];   // P tile staging

    int wid  = blockIdx.x;          // 0..8191
    int qt   = wid & 127;           // query tile within (b,h)
    int bh   = wid >> 7;            // 0..63
    int b    = bh >> 4;
    int h    = bh & 15;
    int lane = threadIdx.x & 31;
    int hi   = lane >> 4;
    int lq   = lane & 15;
    int qbase = qt * 16;

    // Q A-operands (row m = qbase+lq, d contiguous), split d = 0..31 / 32..63
    const _Float16* qrow = qh + ((size_t)b * LL + qbase + lq) * DD + h * DH;
    half16 aq0 = ld2(qrow + hi * 16,      qrow + hi * 16 + 8);
    half16 aq1 = ld2(qrow + 32 + hi * 16, qrow + 40 + hi * 16);

    f32x8 acc[4] = {};
    float mr[8], lr[8];
#pragma unroll
    for (int r = 0; r < 8; ++r) { mr[r] = -3.0e38f; lr[r] = 0.0f; }

    int nkt = (qbase + 16 + 31) >> 5;   // 32-key tiles covering keys <= qbase+15
    for (int j = 0; j < nkt; ++j) {
        int kb = j << 5;

        // --- S = Q @ K^T (two 16-key subtiles, d accumulated 0..31, 32..63) ---
        const _Float16* krow0 = kh + ((size_t)b * LL + kb + lq) * DD + h * DH;
        const _Float16* krow1 = krow0 + (size_t)16 * DD;
        f32x8 z0 = {};
        f32x8 s0 = wmma_f16(aq0, ld2(krow0 + hi * 16, krow0 + hi * 16 + 8), z0);
        s0 = wmma_f16(aq1, ld2(krow0 + 32 + hi * 16, krow0 + 40 + hi * 16), s0);
        f32x8 s1 = wmma_f16(aq0, ld2(krow1 + hi * 16, krow1 + hi * 16 + 8), z0);
        s1 = wmma_f16(aq1, ld2(krow1 + 32 + hi * 16, krow1 + 40 + hi * 16), s1);

        // --- scale + causal mask + online softmax (per-row r in C layout) ---
        bool need_mask = (kb + 31) > qbase;
        float corr[8];
#pragma unroll
        for (int r = 0; r < 8; ++r) {
            int m = qbase + r + 8 * hi;
            float v0 = s0[r] * SCALE;
            float v1 = s1[r] * SCALE;
            if (need_mask) {
                if (kb + lq      > m) v0 = MASKV;
                if (kb + 16 + lq > m) v1 = MASKV;
            }
            float rm = fmaxf(v0, v1);
#pragma unroll
            for (int off = 8; off; off >>= 1)
                rm = fmaxf(rm, __shfl_xor(rm, off, 32));
            float mn = fmaxf(mr[r], rm);
            float c  = __expf(mr[r] - mn);
            float p0 = __expf(v0 - mn);
            float p1 = __expf(v1 - mn);
            float rs = p0 + p1;
#pragma unroll
            for (int off = 8; off; off >>= 1)
                rs += __shfl_xor(rs, off, 32);
            lr[r] = lr[r] * c + rs;
            mr[r] = mn;
            corr[r] = c;
            s0[r] = p0;
            s1[r] = p1;
        }
#pragma unroll
        for (int sub = 0; sub < 4; ++sub)
#pragma unroll
            for (int r = 0; r < 8; ++r)
                acc[sub][r] *= corr[r];

        // --- C-layout P -> LDS (row major 16x32) -> A-layout reload ---
        __syncthreads();
#pragma unroll
        for (int r = 0; r < 8; ++r) {
            pbuf[(r + 8 * hi) * 32 + lq]      = (_Float16)s0[r];
            pbuf[(r + 8 * hi) * 32 + 16 + lq] = (_Float16)s1[r];
        }
        __syncthreads();
        const _Float16* pr = pbuf + lq * 32;
        half16 ap = ld2(pr + hi * 16, pr + hi * 16 + 8);

        // --- O += P @ V  (V^T layout: [bh][d][n], keys contiguous) ---
#pragma unroll
        for (int sub = 0; sub < 4; ++sub) {
            const _Float16* vcol =
                vt + ((size_t)bh * DH + sub * 16 + lq) * (size_t)LL + kb + hi * 16;
            acc[sub] = wmma_f16(ap, ld2(vcol, vcol + 8), acc[sub]);
        }
    }

    // normalize and store z[b, m, h*64+d]
#pragma unroll
    for (int sub = 0; sub < 4; ++sub)
#pragma unroll
        for (int r = 0; r < 8; ++r) {
            float v = acc[sub][r] / lr[r];
            zh[((size_t)b * LL + qbase + r + 8 * hi) * DD + h * DH + sub * 16 + lq] =
                (_Float16)v;
        }
}

// ---------------------------------------------------------------------------
// out = LayerNorm(relu(oh) + x) * gamma + beta ; one block per row of 1024
__global__ __launch_bounds__(256) void ln_kernel(const _Float16* __restrict__ oh,
                                                 const float* __restrict__ x,
                                                 const float* __restrict__ gamma,
                                                 const float* __restrict__ beta,
                                                 float* __restrict__ out) {
    __shared__ float sbuf[16];
    int row  = blockIdx.x;
    int lane = threadIdx.x & 31;
    int wv   = threadIdx.x >> 5;
    const _Float16* orow = oh + (size_t)row * DD;
    const float*    xrow = x  + (size_t)row * DD;

    float y[4];
    float s = 0.0f, ss = 0.0f;
#pragma unroll
    for (int j = 0; j < 4; ++j) {
        int c = threadIdx.x + j * 256;
        float v = fmaxf((float)orow[c], 0.0f) + xrow[c];
        y[j] = v;
        s  += v;
        ss += v * v;
    }
#pragma unroll
    for (int off = 16; off; off >>= 1) {
        s  += __shfl_xor(s,  off, 32);
        ss += __shfl_xor(ss, off, 32);
    }
    if (lane == 0) { sbuf[wv] = s; sbuf[8 + wv] = ss; }
    __syncthreads();
    if (threadIdx.x == 0) {
        float S = 0.0f, SS = 0.0f;
        for (int w = 0; w < 8; ++w) { S += sbuf[w]; SS += sbuf[8 + w]; }
        sbuf[0] = S; sbuf[8] = SS;
    }
    __syncthreads();
    float mu  = sbuf[0] * (1.0f / DD);
    float var = sbuf[8] * (1.0f / DD) - mu * mu;
    float inv = rsqrtf(var + 1e-12f);
#pragma unroll
    for (int j = 0; j < 4; ++j) {
        int c = threadIdx.x + j * 256;
        out[(size_t)row * DD + c] = (y[j] - mu) * inv * gamma[c] + beta[c];
    }
}

// ---------------------------------------------------------------------------
extern "C" void kernel_launch(void* const* d_in, const int* in_sizes, int n_in,
                              void* d_out, int out_size, void* d_ws, size_t ws_size,
                              hipStream_t stream) {
    const float* x     = (const float*)d_in[0];
    const float* Wq    = (const float*)d_in[1];
    const float* Wk    = (const float*)d_in[2];
    const float* Wv    = (const float*)d_in[3];
    const float* Wo    = (const float*)d_in[4];
    const float* gamma = (const float*)d_in[5];
    const float* beta  = (const float*)d_in[6];
    (void)in_sizes; (void)n_in; (void)out_size;

    const size_t MB = 1u << 20;
    char* ws = (char*)d_ws;
    if (ws_size < 89 * MB) return;   // need ~88 MiB

    _Float16* xh  = (_Float16*)(ws + 0);            // 16 MiB  (aliased by vt later)
    _Float16* wqT = (_Float16*)(ws + 16 * MB);      // 2 MiB
    _Float16* wkT = (_Float16*)(ws + 18 * MB);
    _Float16* wvT = (_Float16*)(ws + 20 * MB);
    _Float16* woT = (_Float16*)(ws + 22 * MB);
    _Float16* qh  = (_Float16*)(ws + 24 * MB);      // 16 MiB (aliased by oh later)
    _Float16* kh  = (_Float16*)(ws + 40 * MB);      // 16 MiB
    _Float16* vh  = (_Float16*)(ws + 56 * MB);      // 16 MiB
    _Float16* zh  = (_Float16*)(ws + 72 * MB);      // 16 MiB
    _Float16* vt  = xh;                             // V^T, reuses xh slot
    _Float16* oh  = qh;                             // final GEMM out, reuses qh slot

    const int nElem = NROW * DD;                    // 8,388,608

    cvt_f16_kernel<<<nElem / 256, 256, 0, stream>>>(x, xh, nElem);
    wtrans_kernel<<<DD, 256, 0, stream>>>(Wq, wqT);
    wtrans_kernel<<<DD, 256, 0, stream>>>(Wk, wkT);
    wtrans_kernel<<<DD, 256, 0, stream>>>(Wv, wvT);
    wtrans_kernel<<<DD, 256, 0, stream>>>(Wo, woT);

    // Q/K/V projections: 8192 waves each = 1024 blocks of 8 waves
    gemm_kernel<<<1024, 256, 0, stream>>>(xh, wqT, qh);
    gemm_kernel<<<1024, 256, 0, stream>>>(xh, wkT, kh);
    gemm_kernel<<<1024, 256, 0, stream>>>(xh, wvT, vh);

    repackv_kernel<<<nElem / 256, 256, 0, stream>>>(vh, vt);

    // 8192 query tiles, one wave each
    attn_kernel<<<BB * HH * (LL / 16), 32, 0, stream>>>(qh, kh, vt, zh);

    // output projection
    gemm_kernel<<<1024, 256, 0, stream>>>(zh, woT, oh);

    // relu + residual + layernorm
    ln_kernel<<<NROW, 256, 0, stream>>>(oh, x, gamma, beta, (float*)d_out);
}
